// instanceNormEntropyCNN_11905649344665
// MI455X (gfx1250) — compile-verified, matrix operand order checked by reading, not run on hardware
//
#include <hip/hip_runtime.h>
#include <hip/hip_bf16.h>

typedef __attribute__((ext_vector_type(16))) _Float16 v16h;
typedef __attribute__((ext_vector_type(8)))  float    v8f;
typedef __attribute__((ext_vector_type(4)))  unsigned int u32x4;
typedef __attribute__((ext_vector_type(8)))  int      i32x8;
typedef __attribute__((ext_vector_type(4)))  int      i32x4;

// ---------------------------------------------------------------------------
// TDM: 1-D contiguous DMA of `nelem` f16 elements global -> LDS.
// D# per CDNA5 ISA (group0: count/lds/global/type,
// group1: data_size=2B, tensor_dim0 = tile_dim0 = nelem, stride0 = nelem).
// clang-23 6-arg builtin form: (u32x4, i32x8, i32x4, i32x4, i32x8, cpol).
// Issue from ONE wave; pair with s_wait_tensorcnt(0) + workgroup barrier
// before consuming LDS.
// ---------------------------------------------------------------------------
__device__ inline void tdm_load_f16_1d(void* lds_dst, const void* gsrc,
                                       unsigned nelem) {
  // flat shared-aperture address: low 32 bits == LDS byte offset (ISA §10.2)
  unsigned lds = (unsigned)(unsigned long long)(uintptr_t)lds_dst;
  unsigned long long ga = (unsigned long long)(uintptr_t)gsrc;

  u32x4 g0;
  g0[0] = 1u;                                            // count=1 (valid), load
  g0[1] = lds;                                           // lds_addr (bytes)
  g0[2] = (unsigned)ga;                                  // global_addr[31:0]
  g0[3] = (unsigned)((ga >> 32) & 0x1FFFFFFu) | (2u << 30); // ga[56:32] | type=2

  i32x8 g1;
  g1[0] = (int)(1u << 16);                               // data_size = 2 bytes
  g1[1] = (int)((nelem & 0xFFFFu) << 16);                // tensor_dim0[15:0]
  g1[2] = (int)(((nelem >> 16) & 0xFFFFu) | (1u << 16)); // td0 hi | tensor_dim1=1
  g1[3] = (int)((nelem & 0xFFFFu) << 16);                // tile_dim0 = nelem
  g1[4] = 0;                                             // tile_dim1/2 unused
  g1[5] = (int)nelem;                                    // tensor_dim0_stride lo
  g1[6] = 0;
  g1[7] = 0;

  i32x4 gz4 = {0, 0, 0, 0};
  i32x8 gz8 = {0, 0, 0, 0, 0, 0, 0, 0};
  __builtin_amdgcn_tensor_load_to_lds(g0, g1, gz4, gz4, gz8, 0);
}

// ---------------------------------------------------------------------------
// f32 -> f16 elementwise convert (inputs and weights)
// ---------------------------------------------------------------------------
__global__ __launch_bounds__(256) void cvt_f32_f16(const float* __restrict__ in,
                                                   _Float16* __restrict__ out, int n) {
  int i = blockIdx.x * 256 + threadIdx.x;
  if (i < n) out[i] = (_Float16)in[i];
}

// ---------------------------------------------------------------------------
// Direct-gather implicit-GEMM conv on v_wmma_f32_16x16x32_f16 (used for conv0,
// whose per-image input is too large to stage without killing occupancy).
// One wave = one 16x16 output tile. Fragment layouts per ISA 7.12.2.
// ---------------------------------------------------------------------------
template<int CIN,int HIN,int WIN,int COUT,int KH,int KW,int STR,int PAD,int HOUT,int WOUT>
__global__ __launch_bounds__(32) void conv_wmma(const _Float16* __restrict__ in,
                                                const _Float16* __restrict__ wgt,
                                                const float* __restrict__ bias,
                                                float* __restrict__ out) {
  constexpr int M = HOUT * WOUT;
  constexpr int K = CIN * KH * KW;
  constexpr int TILES_M = (M + 15) / 16;
  constexpr int TILES_N = (COUT + 15) / 16;

  int bid = blockIdx.x;
  int img = bid / (TILES_M * TILES_N);
  int t   = bid % (TILES_M * TILES_N);
  int tm  = t / TILES_N;
  int tn  = t % TILES_N;
  int lane = threadIdx.x;

  const _Float16* inb = in + (size_t)img * CIN * HIN * WIN;

  int mrow = tm * 16 + (lane & 15);
  int oy = mrow / WOUT, ox = mrow % WOUT;
  int ncol = tn * 16 + (lane & 15);

  v8f acc = {};
  for (int k0 = 0; k0 < K; k0 += 32) {
    __builtin_prefetch(wgt + (size_t)(lane & 15) * K + k0 + 32, 0, 1);
    v16h a, b;
#pragma unroll
    for (int i = 0; i < 16; ++i) {
      int kk = k0 + i + ((lane >= 16) ? 8 : 0) + ((i >= 8) ? 8 : 0);
      _Float16 av = (_Float16)0.f, bv = (_Float16)0.f;
      if (kk < K) {
        int c  = kk / (KH * KW);
        int r  = kk % (KH * KW);
        int fy = r / KW, fx = r % KW;
        if (mrow < M) {
          int iy = oy * STR - PAD + fy;
          int ix = ox * STR - PAD + fx;
          if (iy >= 0 && iy < HIN && ix >= 0 && ix < WIN)
            av = inb[(size_t)c * HIN * WIN + (size_t)iy * WIN + ix];
        }
        if (ncol < COUT)
          bv = wgt[(size_t)ncol * K + kk];
      }
      a[i] = av;
      b[i] = bv;
    }
    acc = __builtin_amdgcn_wmma_f32_16x16x32_f16(false, a, false, b,
                                                 (short)0, acc, false, false);
  }

  float* outb = out + (size_t)img * COUT * M;
#pragma unroll
  for (int r = 0; r < 8; ++r) {
    int m = tm * 16 + r + ((lane >= 16) ? 8 : 0);
    int n = tn * 16 + (lane & 15);
    if (m < M && n < COUT)
      outb[(size_t)n * M + m] = acc[r] + bias[n];
  }
}

// ---------------------------------------------------------------------------
// TDM-staged implicit-GEMM conv: one workgroup (8 waves) per image.
// Wave 0 DMAs the whole input plane + f16 weights into LDS via
// tensor_load_to_lds, waits on TENSORcnt, then all waves gather fragments
// from LDS and issue WMMAs over the image's output tiles.
// ---------------------------------------------------------------------------
template<int CIN,int HIN,int WIN,int COUT,int KH,int KW,int STR,int PAD,int HOUT,int WOUT>
__global__ __launch_bounds__(256) void conv_wmma_lds(const _Float16* __restrict__ in,
                                                     const _Float16* __restrict__ wgt,
                                                     const float* __restrict__ bias,
                                                     float* __restrict__ out) {
  constexpr int M = HOUT * WOUT;
  constexpr int K = CIN * KH * KW;
  constexpr int TILES_M = (M + 15) / 16;
  constexpr int TILES_N = (COUT + 15) / 16;
  constexpr int NIN = CIN * HIN * WIN;
  constexpr int NW  = COUT * K;

  __shared__ _Float16 sIn[NIN];
  __shared__ _Float16 sW[NW];

  int img = blockIdx.x;
  int tid = threadIdx.x;

  if (tid < 32) {  // wave 0 issues the two DMA descriptors
    tdm_load_f16_1d(sIn, in + (size_t)img * NIN, (unsigned)NIN);
    tdm_load_f16_1d(sW, wgt, (unsigned)NW);
    __builtin_amdgcn_s_wait_tensorcnt(0);
  }
  __syncthreads();

  int wave = tid >> 5, lane = tid & 31;
  float* outb = out + (size_t)img * COUT * M;

  for (int t = wave; t < TILES_M * TILES_N; t += 8) {
    int tm = t / TILES_N, tn = t % TILES_N;
    int mrow = tm * 16 + (lane & 15);
    int oy = mrow / WOUT, ox = mrow % WOUT;
    int ncol = tn * 16 + (lane & 15);

    v8f acc = {};
    for (int k0 = 0; k0 < K; k0 += 32) {
      v16h a, b;
#pragma unroll
      for (int i = 0; i < 16; ++i) {
        int kk = k0 + i + ((lane >= 16) ? 8 : 0) + ((i >= 8) ? 8 : 0);
        _Float16 av = (_Float16)0.f, bv = (_Float16)0.f;
        if (kk < K) {
          int c  = kk / (KH * KW);
          int r  = kk % (KH * KW);
          int fy = r / KW, fx = r % KW;
          if (mrow < M) {
            int iy = oy * STR - PAD + fy;
            int ix = ox * STR - PAD + fx;
            if (iy >= 0 && iy < HIN && ix >= 0 && ix < WIN)
              av = sIn[c * HIN * WIN + iy * WIN + ix];
          }
          if (ncol < COUT)
            bv = sW[ncol * K + kk];
        }
        a[i] = av;
        b[i] = bv;
      }
      acc = __builtin_amdgcn_wmma_f32_16x16x32_f16(false, a, false, b,
                                                   (short)0, acc, false, false);
    }
#pragma unroll
    for (int r = 0; r < 8; ++r) {
      int m = tm * 16 + r + ((lane >= 16) ? 8 : 0);
      int n = tn * 16 + (lane & 15);
      if (m < M && n < COUT)
        outb[(size_t)n * M + m] = acc[r] + bias[n];
    }
  }
}

// ---------------------------------------------------------------------------
// Per-(image,channel) GroupNorm (biased var) + ReLU, emit f16 for next conv.
// ---------------------------------------------------------------------------
__global__ __launch_bounds__(256) void gn_relu_f16(const float* __restrict__ in,
                                                   const float* __restrict__ w,
                                                   const float* __restrict__ b,
                                                   _Float16* __restrict__ out,
                                                   int HW, int C) {
  int img = blockIdx.x / C, c = blockIdx.x % C;
  const float* p = in + ((size_t)img * C + c) * HW;
  __shared__ float s1[256], s2[256];
  int tid = threadIdx.x;
  float sum = 0.f, sq = 0.f;
  for (int i = tid; i < HW; i += 256) { float v = p[i]; sum += v; sq += v * v; }
  s1[tid] = sum; s2[tid] = sq; __syncthreads();
  for (int off = 128; off > 0; off >>= 1) {
    if (tid < off) { s1[tid] += s1[tid + off]; s2[tid] += s2[tid + off]; }
    __syncthreads();
  }
  float mu  = s1[0] / HW;
  float var = s2[0] / HW - mu * mu;
  float inv = rsqrtf(var + 1e-5f);
  float sc  = w[c] * inv;
  float bb  = b[c] - mu * sc;
  _Float16* q = out + ((size_t)img * C + c) * HW;
  for (int i = tid; i < HW; i += 256) {
    float v = fmaxf(p[i] * sc + bb, 0.f);
    q[i] = (_Float16)v;
  }
}

// ---------------------------------------------------------------------------
// Entropy-scaled InstanceNorm (unbiased var) + ReLU over an 18x18 map.
// Per-patch 25-bin histogram via static-index compare reduction (stays in
// VGPRs, no scratch).
// ---------------------------------------------------------------------------
__global__ __launch_bounds__(256) void inorm_ent(const float* __restrict__ in,
                                                 const float* __restrict__ w,
                                                 const float* __restrict__ b,
                                                 _Float16* __restrict__ outh,
                                                 float* __restrict__ outf,
                                                 int C) {
  const int H = 18, W = 18, HW = 324; (void)H;
  int img = blockIdx.x / C, c = blockIdx.x % C;
  const float* p = in + ((size_t)img * C + c) * HW;
  __shared__ float tile[HW];
  __shared__ float s1[256], s2[256];
  int tid = threadIdx.x;

  float sum = 0.f, sq = 0.f;
  for (int i = tid; i < HW; i += 256) { float v = p[i]; tile[i] = v; sum += v; sq += v * v; }
  s1[tid] = sum; s2[tid] = sq; __syncthreads();
  for (int off = 128; off > 0; off >>= 1) {
    if (tid < off) { s1[tid] += s1[tid + off]; s2[tid] += s2[tid + off]; }
    __syncthreads();
  }
  float mu  = s1[0] / HW;
  float var = (s2[0] / HW - mu * mu) * ((float)HW / (float)(HW - 1)); // unbiased
  __syncthreads();

  float esum = 0.f;
  for (int pidx = tid; pidx < 196; pidx += 256) {
    int py = pidx / 14, px = pidx % 14;
    float vals[25];
#pragma unroll
    for (int e = 0; e < 25; ++e)
      vals[e] = tile[(py + e / 5) * W + px + (e % 5)];
    float mn = vals[0], mx = vals[0];
#pragma unroll
    for (int e = 1; e < 25; ++e) { mn = fminf(mn, vals[e]); mx = fmaxf(mx, vals[e]); }
    float rng = (mx > mn) ? (mx - mn) : 1.f;
    float inv25 = 25.f / rng;
    int bins[25];
#pragma unroll
    for (int e = 0; e < 25; ++e) {
      int bi = (int)floorf((vals[e] - mn) * inv25);
      bins[e] = bi < 0 ? 0 : (bi > 24 ? 24 : bi);
    }
    float ent = 0.f;
#pragma unroll
    for (int bb = 0; bb < 25; ++bb) {
      int cnt = 0;
#pragma unroll
      for (int e = 0; e < 25; ++e) cnt += (bins[e] == bb) ? 1 : 0;
      if (cnt) {
        float pk = (float)cnt * (1.f / 25.f);
        float pc = fminf(fmaxf(pk, 1e-5f), 1.f - 1e-5f);
        ent -= pk * logf(pc);
      }
    }
    esum += ent;
  }
  s1[tid] = esum; __syncthreads();
  for (int off = 128; off > 0; off >>= 1) {
    if (tid < off) s1[tid] += s1[tid + off];
    __syncthreads();
  }
  float entm = s1[0] / 196.f;

  float inv = rsqrtf(var + 1e-5f);
  float sc  = w[c] * inv * entm;
  float bb2 = b[c] - mu * sc;
  size_t base = ((size_t)img * C + c) * HW;
  for (int i = tid; i < HW; i += 256) {
    float v = fmaxf(tile[i] * sc + bb2, 0.f);
    if (outh) outh[base + i] = (_Float16)v;
    if (outf) outf[base + i] = v;
  }
}

// ---------------------------------------------------------------------------
// BatchNorm1d over the batch dim (64 x 10368), biased var; emits f16 for fc1.
// ---------------------------------------------------------------------------
__global__ __launch_bounds__(256) void bn1d_f16(const float* __restrict__ h,
                                                const float* __restrict__ g,
                                                const float* __restrict__ b,
                                                _Float16* __restrict__ out) {
  const int F = 10368, B = 64;
  int f = blockIdx.x * 256 + threadIdx.x;
  if (f >= F) return;
  float sum = 0.f, sq = 0.f;
  for (int i = 0; i < B; ++i) { float v = h[(size_t)i * F + f]; sum += v; sq += v * v; }
  float mu  = sum / B;
  float var = sq / B - mu * mu;
  float inv = rsqrtf(var + 1e-5f);
  float sc  = g[f] * inv;
  float bb  = b[f] - mu * sc;
  for (int i = 0; i < B; ++i)
    out[(size_t)i * F + f] = (_Float16)(h[(size_t)i * F + f] * sc + bb);
}

// ---------------------------------------------------------------------------
// fc1: (64 x 10368) @ (10368 x 32) + bias, ReLU. 8 waves, WMMA-aligned dims.
// ---------------------------------------------------------------------------
__global__ __launch_bounds__(32) void fc1_wmma(const _Float16* __restrict__ A,
                                               const _Float16* __restrict__ Wt,
                                               const float* __restrict__ bias,
                                               float* __restrict__ out) {
  const int Kd = 10368;
  int tm = blockIdx.x / 2, tn = blockIdx.x % 2;
  int lane = threadIdx.x;
  int mrow = tm * 16 + (lane & 15);
  int ncol = tn * 16 + (lane & 15);
  v8f acc = {};
  for (int k0 = 0; k0 < Kd; k0 += 32) {
    __builtin_prefetch(A + (size_t)mrow * Kd + k0 + 64, 0, 1);
    v16h a, b;
#pragma unroll
    for (int i = 0; i < 16; ++i) {
      int kk = k0 + i + ((lane >= 16) ? 8 : 0) + ((i >= 8) ? 8 : 0);
      a[i] = A [(size_t)mrow * Kd + kk];
      b[i] = Wt[(size_t)ncol * Kd + kk];
    }
    acc = __builtin_amdgcn_wmma_f32_16x16x32_f16(false, a, false, b,
                                                 (short)0, acc, false, false);
  }
#pragma unroll
  for (int r = 0; r < 8; ++r) {
    int m = tm * 16 + r + ((lane >= 16) ? 8 : 0);
    int n = tn * 16 + (lane & 15);
    out[m * 32 + n] = fmaxf(acc[r] + bias[n], 0.f);
  }
}

// ---------------------------------------------------------------------------
// Heads: two tiny GEMVs per row + softmax. One thread per batch row.
// ---------------------------------------------------------------------------
__global__ __launch_bounds__(64) void heads(const float* __restrict__ h,
                                            const float* __restrict__ sw,
                                            const float* __restrict__ sb,
                                            const float* __restrict__ vw,
                                            const float* __restrict__ vb,
                                            float* __restrict__ out) {
  int row = threadIdx.x;
  if (row >= 64) return;
  const float* hr = h + row * 32;

  float l5[5]; float mx = -1e30f;
#pragma unroll
  for (int o = 0; o < 5; ++o) {
    float s = sb[o];
    for (int k = 0; k < 32; ++k) s += sw[o * 32 + k] * hr[k];
    l5[o] = s; mx = fmaxf(mx, s);
  }
  float den = 0.f;
#pragma unroll
  for (int o = 0; o < 5; ++o) { l5[o] = expf(l5[o] - mx); den += l5[o]; }
#pragma unroll
  for (int o = 0; o < 5; ++o) out[row * 5 + o] = l5[o] / den;

  float l2[2]; mx = -1e30f;
#pragma unroll
  for (int o = 0; o < 2; ++o) {
    float s = vb[o];
    for (int k = 0; k < 32; ++k) s += vw[o * 32 + k] * hr[k];
    l2[o] = s; mx = fmaxf(mx, s);
  }
  den = 0.f;
#pragma unroll
  for (int o = 0; o < 2; ++o) { l2[o] = expf(l2[o] - mx); den += l2[o]; }
  float* vout = out + 64 * 5;
#pragma unroll
  for (int o = 0; o < 2; ++o) vout[row * 2 + o] = l2[o] / den;
}

// ---------------------------------------------------------------------------
extern "C" void kernel_launch(void* const* d_in, const int* in_sizes, int n_in,
                              void* d_out, int out_size, void* d_ws, size_t ws_size,
                              hipStream_t stream) {
  const float* x    = (const float*)d_in[0];
  const float* c0w  = (const float*)d_in[1];
  const float* c0b  = (const float*)d_in[2];
  const float* c1w  = (const float*)d_in[3];
  const float* c1b  = (const float*)d_in[4];
  const float* c2w  = (const float*)d_in[5];
  const float* c2b  = (const float*)d_in[6];
  const float* gn0w = (const float*)d_in[7];
  const float* gn0b = (const float*)d_in[8];
  const float* n1w  = (const float*)d_in[9];
  const float* n1b  = (const float*)d_in[10];
  const float* n2w  = (const float*)d_in[11];
  const float* n2b  = (const float*)d_in[12];
  const float* bng  = (const float*)d_in[13];
  const float* bnb  = (const float*)d_in[14];
  const float* fc1w = (const float*)d_in[15];
  const float* fc1b = (const float*)d_in[16];
  const float* shw  = (const float*)d_in[17];
  const float* shb  = (const float*)d_in[18];
  const float* vnw  = (const float*)d_in[19];
  const float* vnb  = (const float*)d_in[20];
  float* out = (float*)d_out;

  char* ws = (char*)d_ws;
  size_t off = 0;
  auto carve = [&](size_t bytes) -> char* {
    char* p = ws + off;
    off = (off + bytes + 255) & ~(size_t)255;
    return p;
  };
  const int NX   = 64 * 3 * 224 * 224;
  const int NH0  = 64 * 8 * 3136;
  const int NH1  = 64 * 16 * 324;
  const int NH2  = 64 * 32 * 324;
  const int NFCW = 32 * 10368;

  _Float16* xf16 = (_Float16*)carve((size_t)NX * 2);
  _Float16* w0f  = (_Float16*)carve(1176 * 2);
  _Float16* w1f  = (_Float16*)carve(3200 * 2);
  _Float16* w2f  = (_Float16*)carve(4608 * 2);
  _Float16* fcwf = (_Float16*)carve((size_t)NFCW * 2);
  float*    h0   = (float*)   carve((size_t)NH0 * 4);
  _Float16* h0f  = (_Float16*)carve((size_t)NH0 * 2);
  float*    h1   = (float*)   carve((size_t)NH1 * 4);
  _Float16* h1f  = (_Float16*)carve((size_t)NH1 * 2);
  float*    h2   = (float*)   carve((size_t)NH2 * 4);
  float*    h2n  = (float*)   carve((size_t)NH2 * 4);
  _Float16* bnf  = (_Float16*)carve((size_t)64 * 10368 * 2);
  float*    fc1o = (float*)   carve((size_t)64 * 32 * 4);
  (void)ws_size; (void)in_sizes; (void)n_in; (void)out_size;

  cvt_f32_f16<<<(NX + 255) / 256, 256, 0, stream>>>(x, xf16, NX);
  cvt_f32_f16<<<(1176 + 255) / 256, 256, 0, stream>>>(c0w, w0f, 1176);
  cvt_f32_f16<<<(3200 + 255) / 256, 256, 0, stream>>>(c1w, w1f, 3200);
  cvt_f32_f16<<<(4608 + 255) / 256, 256, 0, stream>>>(c2w, w2f, 4608);
  cvt_f32_f16<<<(NFCW + 255) / 256, 256, 0, stream>>>(fc1w, fcwf, NFCW);

  // conv0: direct-gather WMMA (input too big to stage per-WG)
  conv_wmma<3, 224, 224, 8, 7, 7, 4, 2, 56, 56>
      <<<64 * 196 * 1, 32, 0, stream>>>(xf16, w0f, c0b, h0);
  gn_relu_f16<<<64 * 8, 256, 0, stream>>>(h0, gn0w, gn0b, h0f, 3136, 8);

  // conv1: TDM-staged (49KB input + 6.4KB weights in LDS), 1 WG / image
  conv_wmma_lds<8, 56, 56, 16, 5, 5, 3, 1, 18, 18>
      <<<64, 256, 0, stream>>>(h0f, w1f, c1b, h1);
  inorm_ent<<<64 * 16, 256, 0, stream>>>(h1, n1w, n1b, h1f, nullptr, 16);

  // conv2: TDM-staged (10KB input + 9.2KB weights in LDS), 1 WG / image
  conv_wmma_lds<16, 18, 18, 32, 3, 3, 1, 1, 18, 18>
      <<<64, 256, 0, stream>>>(h1f, w2f, c2b, h2);
  inorm_ent<<<64 * 32, 256, 0, stream>>>(h2, n2w, n2b, nullptr, h2n, 32);

  bn1d_f16<<<(10368 + 255) / 256, 256, 0, stream>>>(h2n, bng, bnb, bnf);
  fc1_wmma<<<8, 32, 0, stream>>>(bnf, fcwf, fc1b, fc1o);
  heads<<<1, 64, 0, stream>>>(fc1o, shw, shb, vnw, vnb, out);
}
